// BlockwiseAttention3D_1537598292351
// MI455X (gfx1250) — matrix operand order repaired and checked by reading, not compiled, not run on hardware
//
#include <hip/hip_runtime.h>
#include <hip/hip_bf16.h>

// ---------------------------------------------------------------------------
// BlockwiseAttention3D on MI455X (gfx1250), wave32 + WMMA bf16 + TDM.
//
// Memory-bound (~6.2 GFLOP vs ~256 MB mandatory x/out traffic @ 23.3 TB/s):
//   kernel 1: out = x (float4 copy; 7/8 of voxels are pass-through)
//   kernel 2: per diagonal block p: TENSOR_LOAD_TO_LDS stages the x-tile
//             (32 channels per chunk), convert to bf16 [n][c] in LDS, then
//             Q/K/V projections via v_wmma_f32_16x16x32_bf16.
//   kernel 3: per (p, 64-query tile): S=QK^T (WMMA), softmax in LDS, O=P V^T
//             (WMMA), out = gamma*O + x on diagonal voxels (float4 stores).
// All WMMA fragments are contiguous per lane -> b128 loads; loop structure
// keeps <= 4 B-fragments live so nothing spills to scratch.
// Workspace (bf16): qt[64][512][32] kt[64][512][32] vtT[64][128][512] = 12 MB.
// ---------------------------------------------------------------------------

typedef __attribute__((ext_vector_type(16))) __bf16 v16bf;
typedef __attribute__((ext_vector_type(8)))  __bf16 v8bf;
typedef __attribute__((ext_vector_type(8)))  float  v8f;
typedef __attribute__((ext_vector_type(4)))  unsigned int v4u;
typedef __attribute__((ext_vector_type(8)))  int   v8i;
typedef __attribute__((ext_vector_type(4)))  int   v4i;

#define C_IN      128
#define NVOX      512     // 8*8*8 voxels per block
#define DQK_PAD   32      // q/k channel dim padded 16 -> 32 (zeros)
#define XT_STRIDE 136     // LDS row stride (bf16) for x tile [n][c]; rows 16B-aligned
#define SSTR      514     // LDS row stride (f32) for S (conflict-free row scans)
#define PSTR      520     // LDS row stride (bf16) for P (rows 16B-aligned)

#if defined(__gfx1250__) && __has_builtin(__builtin_amdgcn_tensor_load_to_lds)
#define HAVE_TDM 1
#else
#define HAVE_TDM 0
#endif

__device__ __forceinline__ v8f wmma_bf16(v16bf a, v16bf b, v8f c) {
  return __builtin_amdgcn_wmma_f32_16x16x32_bf16(false, a, false, b, (short)0, c, false, false);
}

// A-matrix 16x32 bf16 fragment from a row-major [16][K] panel (contiguous K):
// lane = half*16 + m ; elements: K = half*16 + e.
__device__ __forceinline__ v16bf load_a_frag(const __bf16* base, int row_stride, int lane) {
  const int l = lane & 15, half = lane >> 4;
  const __bf16* rp = base + l * row_stride + half * 16;
  v8bf lo = *(const v8bf*)rp;
  v8bf hi = *(const v8bf*)(rp + 8);
  return __builtin_shufflevector(lo, hi, 0, 1, 2, 3, 4, 5, 6, 7,
                                 8, 9, 10, 11, 12, 13, 14, 15);
}

// B-matrix 32x16 bf16 fragment from a row-major [16][K] panel (B[k][n] = panel[n][k]):
// lane = half*16 + n ; elements K = half*16 + e. One contiguous 32-byte run.
__device__ __forceinline__ v16bf load_b_frag(const __bf16* base, int row_stride, int lane) {
  const int l = lane & 15, half = lane >> 4;
  return *(const v16bf*)(base + l * row_stride + half * 16);
}

// ---------------------------------------------------------------------------
// Kernel 1: out = x (float4 grid-strided copy)
// ---------------------------------------------------------------------------
__global__ void copy_kernel(const float4* __restrict__ x, float4* __restrict__ out, int n4) {
  int idx = blockIdx.x * blockDim.x + threadIdx.x;
  const int stride = gridDim.x * blockDim.x;
  for (; idx < n4; idx += stride) out[idx] = x[idx];
}

// ---------------------------------------------------------------------------
// TDM: stage one 4D x-chunk (W=8, H=8, D=8, C=c_count) f32 into LDS.
// D# per CDNA5 ISA.
// ---------------------------------------------------------------------------
__device__ __forceinline__ void tdm_load_x_chunk(const float* gsrc, unsigned lds_byte_addr,
                                                 int c_count) {
#if HAVE_TDM
  const unsigned long long ga = (unsigned long long)(uintptr_t)gsrc;
  v4u g0;
  g0[0] = 1u;                                                 // count=1 valid user D#
  g0[1] = lds_byte_addr;                                      // lds_addr
  g0[2] = (unsigned)(ga & 0xffffffffu);                       // global_addr[31:0]
  g0[3] = (unsigned)((ga >> 32) & 0x01ffffffu) | (2u << 30);  // addr[56:32] | type=2
  v8i g1;
  g1[0] = (2 << 16);          // workgroup_mask=0 ; data_size=2 (4 bytes)
  g1[1] = (64 << 16);         // tensor_dim0 (W len = 64) low16 in bits 63:48
  g1[2] = (64 << 16);         // dim0 hi=0 ; tensor_dim1 (H len) low16 = 64
  g1[3] = (8 << 16);          // dim1 hi=0 ; tile_dim0 (W tile) = 8
  g1[4] = 8 | (8 << 16);      // tile_dim1 (H tile)=8 ; tile_dim2 (D tile)=8
  g1[5] = 64;                 // tensor_dim0_stride = 64 (H step, elems)
  g1[6] = (4096 << 16);       // stride0 hi=0 ; tensor_dim1_stride low16 = 4096 (D step)
  g1[7] = 0;                  // tensor_dim1_stride hi
  v4i g2;
  g2[0] = 64;                 // tensor_dim2 (D len)
  g2[1] = c_count;            // tensor_dim3 (C len of this chunk)
  g2[2] = 64 * 64 * 64;       // tensor_dim2_stride (C step, elems)
  g2[3] = (c_count << 16);    // stride2 hi=0 ; tile_dim3 = c_count
  v4i g3;
  g3[0] = 0; g3[1] = 0; g3[2] = 0; g3[3] = 0;
#if __clang_major__ >= 23
  v8i g4 = {0, 0, 0, 0, 0, 0, 0, 0};
  __builtin_amdgcn_tensor_load_to_lds(g0, g1, g2, g3, g4, 0);
#else
  __builtin_amdgcn_tensor_load_to_lds(g0, g1, g2, g3, 0);
#endif
  __builtin_amdgcn_s_wait_tensorcnt(0);
#endif
}

// ---------------------------------------------------------------------------
// Kernel 2: projections. One workgroup (256 thr / 8 waves) per diagonal block p.
// LDS: xt bf16 [n][136] 136KB + stg f32 32ch 64KB + weights 40KB + bias ~ 240KB.
// ---------------------------------------------------------------------------
__global__ void proj_kernel(const float* __restrict__ x,
                            const float* __restrict__ Wq, const float* __restrict__ bq,
                            const float* __restrict__ Wk, const float* __restrict__ bk,
                            const float* __restrict__ Wv, const float* __restrict__ bv,
                            __bf16* __restrict__ qt, __bf16* __restrict__ kt,
                            __bf16* __restrict__ vtT) {
  extern __shared__ char smem_raw[];
  __bf16* xt  = (__bf16*)smem_raw;                  // [n][XT_STRIDE]
  float*  stg = (float*)(xt + NVOX * XT_STRIDE);    // [32][512] f32 TDM chunk
  __bf16* wqs = (__bf16*)(stg + 32 * NVOX);         // [16][128]
  __bf16* wks = wqs + 16 * C_IN;                    // [16][128]
  __bf16* wvs = wks + 16 * C_IN;                    // [128][128]
  float*  bia = (float*)(wvs + C_IN * C_IN);        // 160 floats: bq|bk|bv

  const int p = blockIdx.x;
  const int bi = p >> 3, bj = p & 7;
  const int tid = threadIdx.x;
  const int lane = tid & 31, wave = tid >> 5;
  const int d = lane & 15, half = lane >> 4;

  for (int idx = tid; idx < 16 * C_IN; idx += 256)   wqs[idx] = (__bf16)Wq[idx];
  for (int idx = tid; idx < 16 * C_IN; idx += 256)   wks[idx] = (__bf16)Wk[idx];
  for (int idx = tid; idx < C_IN * C_IN; idx += 256) wvs[idx] = (__bf16)Wv[idx];
  if (tid < 16)      bia[tid] = bq[tid];
  else if (tid < 32) bia[tid] = bk[tid - 16];
  if (tid < C_IN)    bia[32 + tid] = bv[tid];

  // q/k zero padding (columns 16..31), one 32-byte store per row.
  {
    v16bf z = {};
    for (int n = tid; n < NVOX; n += 256) {
      *(v16bf*)(qt + ((size_t)p * NVOX + n) * DQK_PAD + 16) = z;
      *(v16bf*)(kt + ((size_t)p * NVOX + n) * DQK_PAD + 16) = z;
    }
  }

  // Stage x block: 4 TDM chunks of 32 channels each; convert f32 -> bf16 [n][c].
  const float* xblk = x + (size_t)(bi * 8) * 4096 + (size_t)(bj * 8) * 64 + bi * 8;
#if HAVE_TDM
#pragma unroll 1
  for (int ch = 0; ch < 4; ++ch) {
    if (tid < 32) {  // one wave drives the DMA (TDM ignores EXEC; TENSORcnt is per-wave)
      tdm_load_x_chunk(xblk + (size_t)(ch * 32) * 262144, (unsigned)(uintptr_t)stg, 32);
    }
    __syncthreads();
    for (int idx = tid; idx < 32 * NVOX; idx += 256) {
      const int c = idx >> 9, n = idx & (NVOX - 1);
      xt[n * XT_STRIDE + ch * 32 + c] = (__bf16)stg[idx];
    }
    __syncthreads();
  }
#else
  for (int idx = tid; idx < NVOX * C_IN; idx += 256) {
    const int n = idx & (NVOX - 1);
    const int c = idx >> 9;
    const int dd = n >> 6, hh = (n >> 3) & 7, ww = n & 7;
    xt[n * XT_STRIDE + c] =
        (__bf16)x[(((size_t)c * 64 + bi * 8 + dd) * 64 + bj * 8 + hh) * 64 + bi * 8 + ww];
  }
  __syncthreads();
#endif

  // ---- Q projection: hold 4 B-frags (32 VGPRs), stream A from LDS ----
  {
    v16bf b0 = load_b_frag(wqs +  0, C_IN, lane);
    v16bf b1 = load_b_frag(wqs + 32, C_IN, lane);
    v16bf b2 = load_b_frag(wqs + 64, C_IN, lane);
    v16bf b3 = load_b_frag(wqs + 96, C_IN, lane);
    const float bval = bia[d];
#pragma unroll 1
    for (int nt = wave; nt < 32; nt += 8) {
      const __bf16* ab = xt + nt * 16 * XT_STRIDE;
      v8f acc = {};
      acc = wmma_bf16(load_a_frag(ab +  0, XT_STRIDE, lane), b0, acc);
      acc = wmma_bf16(load_a_frag(ab + 32, XT_STRIDE, lane), b1, acc);
      acc = wmma_bf16(load_a_frag(ab + 64, XT_STRIDE, lane), b2, acc);
      acc = wmma_bf16(load_a_frag(ab + 96, XT_STRIDE, lane), b3, acc);
#pragma unroll
      for (int r = 0; r < 8; ++r) {
        const size_t n = (size_t)p * NVOX + nt * 16 + r + 8 * half;
        qt[n * DQK_PAD + d] = (__bf16)(acc[r] + bval);
      }
    }
  }
  // ---- K projection ----
  {
    v16bf b0 = load_b_frag(wks +  0, C_IN, lane);
    v16bf b1 = load_b_frag(wks + 32, C_IN, lane);
    v16bf b2 = load_b_frag(wks + 64, C_IN, lane);
    v16bf b3 = load_b_frag(wks + 96, C_IN, lane);
    const float bval = bia[16 + d];
#pragma unroll 1
    for (int nt = wave; nt < 32; nt += 8) {
      const __bf16* ab = xt + nt * 16 * XT_STRIDE;
      v8f acc = {};
      acc = wmma_bf16(load_a_frag(ab +  0, XT_STRIDE, lane), b0, acc);
      acc = wmma_bf16(load_a_frag(ab + 32, XT_STRIDE, lane), b1, acc);
      acc = wmma_bf16(load_a_frag(ab + 64, XT_STRIDE, lane), b2, acc);
      acc = wmma_bf16(load_a_frag(ab + 96, XT_STRIDE, lane), b3, acc);
#pragma unroll
      for (int r = 0; r < 8; ++r) {
        const size_t n = (size_t)p * NVOX + nt * 16 + r + 8 * half;
        kt[n * DQK_PAD + d] = (__bf16)(acc[r] + bval);
      }
    }
  }
  // ---- V projection: dt outer (4 B-frags live), nt inner streaming A ----
#pragma unroll 1
  for (int dt = 0; dt < 8; ++dt) {
    const __bf16* wb = wvs + dt * 16 * C_IN;
    v16bf b0 = load_b_frag(wb +  0, C_IN, lane);
    v16bf b1 = load_b_frag(wb + 32, C_IN, lane);
    v16bf b2 = load_b_frag(wb + 64, C_IN, lane);
    v16bf b3 = load_b_frag(wb + 96, C_IN, lane);
    const float bval = bia[32 + dt * 16 + d];
#pragma unroll 1
    for (int nt = wave; nt < 32; nt += 8) {
      const __bf16* ab = xt + nt * 16 * XT_STRIDE;
      v8f acc = {};
      acc = wmma_bf16(load_a_frag(ab +  0, XT_STRIDE, lane), b0, acc);
      acc = wmma_bf16(load_a_frag(ab + 32, XT_STRIDE, lane), b1, acc);
      acc = wmma_bf16(load_a_frag(ab + 64, XT_STRIDE, lane), b2, acc);
      acc = wmma_bf16(load_a_frag(ab + 96, XT_STRIDE, lane), b3, acc);
      v8bf o;
#pragma unroll
      for (int r = 0; r < 8; ++r) o[r] = (__bf16)(acc[r] + bval);
      // vtT[p][dt*16+d][nt*16 + 8*half .. +7]: 8 consecutive bf16 -> one b128 store
      *(v8bf*)(vtT + ((size_t)p * C_IN + dt * 16 + d) * NVOX + nt * 16 + 8 * half) = o;
    }
  }
}

// ---------------------------------------------------------------------------
// Kernel 3: attention. One workgroup (256 thr / 8 waves) per (block p, 64-query tile).
// LDS: S f32 (64 x 514) + P bf16 (64 x 520) + 2 reduce buffers ~ 196 KB.
// ---------------------------------------------------------------------------
__global__ void attn_kernel(const float* __restrict__ x, const float* __restrict__ gamma,
                            const __bf16* __restrict__ qt, const __bf16* __restrict__ kt,
                            const __bf16* __restrict__ vtT, float* __restrict__ out) {
  extern __shared__ char smem_raw[];
  float*  S    = (float*)smem_raw;                 // 64 * SSTR
  __bf16* P    = (__bf16*)(S + 64 * SSTR);         // 64 * PSTR
  float*  redM = (float*)(P + 64 * PSTR);          // 256
  float*  redS = redM + 256;                       // 256

  const int p = blockIdx.x >> 3;
  const int qbase = (blockIdx.x & 7) * 64;
  const int bi = p >> 3, bj = p & 7;
  const int tid = threadIdx.x, lane = tid & 31, wave = tid >> 5;
  const int l = lane & 15, half = lane >> 4;

  const __bf16* qtp = qt + (size_t)p * NVOX * DQK_PAD;
  const __bf16* ktp = kt + (size_t)p * NVOX * DQK_PAD;
  const __bf16* vtp = vtT + (size_t)p * C_IN * NVOX;

  // Phase 1: S[64][512] = Q K^T ; A fragment reused across the wave's 4 key tiles.
#pragma unroll 1
  for (int mt = 0; mt < 4; ++mt) {
    v16bf a = load_a_frag(qtp + (size_t)(qbase + mt * 16) * DQK_PAD, DQK_PAD, lane);
#pragma unroll 1
    for (int nt = wave; nt < 32; nt += 8) {
      v16bf b = load_b_frag(ktp + (size_t)(nt * 16) * DQK_PAD, DQK_PAD, lane);
      v8f acc = {};
      acc = wmma_bf16(a, b, acc);
#pragma unroll
      for (int r = 0; r < 8; ++r)
        S[(mt * 16 + r + 8 * half) * SSTR + nt * 16 + l] = acc[r];
    }
  }
  __syncthreads();

  // Phase 2: row softmax, 4 threads per row (all 256 threads active; no WMMA here).
  {
    const int row = tid >> 2, part = tid & 3;
    float* rowp = S + row * SSTR;
    const int c0 = part * 128, c1 = c0 + 128;
    float mx = rowp[c0];
    for (int m = c0 + 1; m < c1; ++m) mx = fmaxf(mx, rowp[m]);
    redM[tid] = mx;
    __syncthreads();
    mx = fmaxf(fmaxf(redM[row * 4], redM[row * 4 + 1]),
               fmaxf(redM[row * 4 + 2], redM[row * 4 + 3]));
    float sum = 0.0f;
    for (int m = c0; m < c1; ++m) { const float e = __expf(rowp[m] - mx); rowp[m] = e; sum += e; }
    redS[tid] = sum;
    __syncthreads();
    sum = redS[row * 4] + redS[row * 4 + 1] + redS[row * 4 + 2] + redS[row * 4 + 3];
    const float inv = 1.0f / sum;
    __bf16* prow = P + row * PSTR;
    for (int m = c0; m < c1; ++m) prow[m] = (__bf16)(rowp[m] * inv);
  }
  __syncthreads();

  const float g = gamma[0];

  // Phase 3: O[64][128] = P @ V^T ; wave = channel tile, 4 query sub-tiles each.
  const int dt = wave;  // 8 waves == 8 channel tiles
#pragma unroll 1
  for (int mt = 0; mt < 4; ++mt) {
    v8f acc = {};
#pragma unroll 4
    for (int kk = 0; kk < 16; ++kk) {
      v16bf a = load_a_frag(P + mt * 16 * PSTR + kk * 32, PSTR, lane);
      // B[k=m][n=d] = vtT[dt*16+n][kk*32 + k] : contiguous per lane
      v16bf b = load_b_frag(vtp + (size_t)(dt * 16) * NVOX + kk * 32, NVOX, lane);
      acc = wmma_bf16(a, b, acc);
    }
    // Epilogue: for fixed channel d the 8 rows are 8 consecutive W-voxels.
    const int d = dt * 16 + l;
    const int n0 = qbase + mt * 16 + 8 * half;          // multiple of 8 -> ww = 0
    const int dd = n0 >> 6, hh = (n0 >> 3) & 7;
    const size_t idx0 = (((size_t)d * 64 + bi * 8 + dd) * 64 + bj * 8 + hh) * 64 + bi * 8;
    float4 xa = *(const float4*)(x + idx0);
    float4 xb = *(const float4*)(x + idx0 + 4);
    float4 oa, ob;
    oa.x = g * acc[0] + xa.x;  oa.y = g * acc[1] + xa.y;
    oa.z = g * acc[2] + xa.z;  oa.w = g * acc[3] + xa.w;
    ob.x = g * acc[4] + xb.x;  ob.y = g * acc[5] + xb.y;
    ob.z = g * acc[6] + xb.z;  ob.w = g * acc[7] + xb.w;
    *(float4*)(out + idx0)     = oa;
    *(float4*)(out + idx0 + 4) = ob;
  }
}

// ---------------------------------------------------------------------------
extern "C" void kernel_launch(void* const* d_in, const int* in_sizes, int n_in,
                              void* d_out, int out_size, void* d_ws, size_t ws_size,
                              hipStream_t stream) {
  const float* x     = (const float*)d_in[0];
  const float* Wq    = (const float*)d_in[1];
  const float* bq    = (const float*)d_in[2];
  const float* Wk    = (const float*)d_in[3];
  const float* bk    = (const float*)d_in[4];
  const float* Wv    = (const float*)d_in[5];
  const float* bv    = (const float*)d_in[6];
  const float* gamma = (const float*)d_in[7];
  float* out = (float*)d_out;

  // Workspace (bf16): qt[64*512*32] kt[64*512*32] vtT[64*128*512] = 12 MB
  __bf16* qt  = (__bf16*)d_ws;
  __bf16* kt  = qt + (size_t)64 * NVOX * DQK_PAD;
  __bf16* vtT = kt + (size_t)64 * NVOX * DQK_PAD;

  const int n4 = (C_IN * 64 * 64 * 64) / 4;
  copy_kernel<<<2048, 256, 0, stream>>>((const float4*)x, (float4*)out, n4);

  const size_t shA = (size_t)NVOX * XT_STRIDE * 2        // xt bf16 [n][c]
                   + (size_t)32 * NVOX * 4               // f32 TDM staging chunk
                   + (size_t)(16 + 16 + C_IN) * C_IN * 2 // weights bf16
                   + 160 * 4;                            // biases f32
  proj_kernel<<<64, 256, shA, stream>>>(x, Wq, bq, Wk, bk, Wv, bv, qt, kt, vtT);

  const size_t shB = (size_t)64 * SSTR * 4 + (size_t)64 * PSTR * 2 + 512 * 4;
  attn_kernel<<<512, 256, shB, stream>>>(x, gamma, qt, kt, vtT, out);
}
